// RNNModel_52621939310793
// MI455X (gfx1250) — matrix operand order repaired and checked by reading, not compiled, not run on hardware
//
#include <hip/hip_runtime.h>
#include <hip/hip_bf16.h>

#define SEQ     128
#define BATCH   32
#define NTOKEN  33278
#define IN_SIZE 512
#define MAXF    0.875f

typedef __attribute__((ext_vector_type(16))) _Float16 v16h;
typedef __attribute__((ext_vector_type(8)))  _Float16 v8h;
typedef __attribute__((ext_vector_type(8)))  float    v8f;
typedef __attribute__((ext_vector_type(4)))  unsigned u32x4;
typedef __attribute__((ext_vector_type(8)))  unsigned u32x8;

// ---------------------------------------------------------------------------
// Fragment load helper (CDNA5 WMMA VGPR layouts, ISA 05_wmma §7.12.2)
// ---------------------------------------------------------------------------
__device__ __forceinline__ v16h ld16_pair(const _Float16* p0, const _Float16* p1) {
    const v8h a = *(const v8h*)p0;
    const v8h b = *(const v8h*)p1;
    v16h r;
#pragma unroll
    for (int i = 0; i < 8; ++i) { r[i] = a[i]; r[i + 8] = b[i]; }
    return r;
}

__device__ __forceinline__ float dev_sigmoid(float x) {
    return 1.0f / (1.0f + __expf(-x));
}

// ---------------------------------------------------------------------------
// Tensor Data Mover: async DMA of a 2D f16 tile (rows x 32 cols) into LDS.
// Builds a D# (groups 0+1, 2D form: VADDR2/3 NULL) per ISA 08_async_tensor §8.
// Tracked by TENSORcnt; caller pairs with __builtin_amdgcn_s_wait_tensorcnt.
// ---------------------------------------------------------------------------
__device__ __forceinline__ void tdm_load_b_tile(const _Float16* gsrc, unsigned lds_byte,
                                                int dim0_rem, int dim1_rem) {
    const unsigned long long ga = (unsigned long long)(uintptr_t)gsrc;
    // group 0: count=1 | lds_addr | global_addr[56:0] | type=2 ("image")
    const u32x4 g0 = {
        1u,
        lds_byte,
        (unsigned)ga,
        ((unsigned)(ga >> 32) & 0x01FFFFFFu) | (2u << 30)
    };
    // group 1: data_size=2B; tensor_dim0/1 (OOB clamp -> zero-fill);
    //          tile_dim0=32, tile_dim1=128; tensor_dim0_stride=512
    const unsigned d0 = (unsigned)dim0_rem;           // elements along K
    const unsigned d1 = (unsigned)dim1_rem;           // rows (vocab)
    const u32x8 g1 = {
        0x00010000u,                                  // [17:16]=data_size=1 (2 bytes)
        (d0 & 0xFFFFu) << 16,                         // tensor_dim0[15:0]  @bits 63:48
        ((d0 >> 16) & 0xFFFFu) | ((d1 & 0xFFFFu) << 16), // dim0[31:16] | dim1[15:0]
        ((d1 >> 16) & 0xFFFFu) | (32u << 16),         // dim1[31:16] | tile_dim0=32
        128u,                                         // tile_dim1=128, tile_dim2=0
        (unsigned)IN_SIZE,                            // tensor_dim0_stride[31:0]=512
        0u,                                           // stride0[47:32] | stride1[15:0]
        0u                                            // tensor_dim1_stride[47:16]
    };
    asm volatile("tensor_load_to_lds %0, %1" :: "s"(g0), "s"(g1) : "memory");
}

// ---------------------------------------------------------------------------
// Prep kernels
// ---------------------------------------------------------------------------
__global__ void cvt_f32_to_f16(const float* __restrict__ src,
                               _Float16* __restrict__ dst, int n) {
    int i = blockIdx.x * blockDim.x + threadIdx.x;
    if (i < n) dst[i] = (_Float16)src[i];
}

// W (K x N, f32, row-major) -> Wt (N x K, f16, row-major)
__global__ void transpose_f32_to_f16(const float* __restrict__ W,
                                     _Float16* __restrict__ Wt, int K, int N) {
    int i = blockIdx.x * blockDim.x + threadIdx.x;
    if (i < K * N) {
        int k = i / N, n = i % N;
        Wt[(size_t)n * K + k] = (_Float16)W[i];
    }
}

__global__ void copy_f32(const float* __restrict__ src, float* __restrict__ dst, int n) {
    int i = blockIdx.x * blockDim.x + threadIdx.x;
    if (i < n) dst[i] = src[i];
}

// x0_h[row, :] = embed_h[seq[row], :]   (row = t*BATCH + b)
__global__ void gather_rows(const int* __restrict__ seq,
                            const _Float16* __restrict__ embed_h,
                            _Float16* __restrict__ xh) {
    const int row = blockIdx.x;
    const int tok = seq[row];
    for (int i = threadIdx.x; i < IN_SIZE; i += blockDim.x)
        xh[(size_t)row * IN_SIZE + i] = embed_h[(size_t)tok * IN_SIZE + i];
}

// ---------------------------------------------------------------------------
// One WMMA sub-GEMM inside a fused RevGRU step.
// A = [lds_x (32 x ISZ) | lds_h (32 x HALF)]  (f16, row-major in LDS)
// Wt = N x K (f16, row-major, K = ISZ+HALF)   (L2-resident)
// out: lds_o[:, col0 .. col0+N) + bias (f32, ldc = 2*HALF)
// All tile counts are multiples of nwaves -> EXEC is all-ones at every WMMA.
// ---------------------------------------------------------------------------
__device__ __forceinline__ void wmma_subgemm(
    const _Float16* lds_x, const _Float16* lds_h,
    int ISZ, int HALF,
    const _Float16* __restrict__ Wt, const float* __restrict__ bias,
    float* lds_o, int col0, int N,
    int lane, int wave, int nwaves)
{
    const int K   = ISZ + HALF;
    const int H2  = 2 * HALF;
    const int l15 = lane & 15;
    const int lhi = lane >> 4;
    const int ntiles = 2 * (N >> 4);            // 2 M-tiles x N/16 N-tiles

    for (int tile = wave; tile < ntiles; tile += nwaves) {
        const int mt = tile & 1;
        const int nt = tile >> 1;
        const int mrow = mt * 16 + l15;
        v8f acc = {};
        for (int k0 = 0; k0 < K; k0 += 32) {
            // A fragment: lane<16 -> K = k0..k0+15 ; lane>=16 -> K = k0+16..k0+31
            const _Float16* ab = (k0 < ISZ)
                ? (lds_x + (size_t)mrow * ISZ + k0)
                : (lds_h + (size_t)mrow * HALF + (k0 - ISZ));
            const _Float16* ap = ab + (lhi << 4);
            v16h a = ld16_pair(ap, ap + 8);
            // B fragment: lane holds column n = l15, K chunk selected by lhi
            const _Float16* bp = Wt + (size_t)(nt * 16 + l15) * K + k0 + (lhi << 4);
            v16h b = ld16_pair(bp, bp + 8);
            acc = __builtin_amdgcn_wmma_f32_16x16x32_f16(
                false, a, false, b, (short)0, acc, false, false);
        }
        const int ncol = nt * 16 + l15;
        const float bv = bias[ncol];
        const int rbase = mt * 16 + (lhi << 3);
#pragma unroll
        for (int j = 0; j < 8; ++j)                    // C layout: VGPR j -> M = j (+8 hi half)
            lds_o[(size_t)(rbase + j) * H2 + col0 + ncol] = acc[j] + bv;
    }
}

// ---------------------------------------------------------------------------
// Fused RevGRU step: one workgroup (32 wave32s) per (layer, t).
// h state (f32) in global ws; activations staged in LDS.
// ---------------------------------------------------------------------------
template <int ISZ, int HALF>
__global__ __launch_bounds__(1024) void revgru_step(
    const _Float16* __restrict__ x_h,       // 32 x ISZ  (f16)
    float* __restrict__ h,                  // 32 x 2*HALF (f32 state, updated in place)
    _Float16* __restrict__ y_h,             // 32 x 2*HALF (f16 output for this t)
    const _Float16* __restrict__ Wzr1t, const float* __restrict__ bzr1,
    const _Float16* __restrict__ Wg1t,  const float* __restrict__ bg1,
    const _Float16* __restrict__ Wzr2t, const float* __restrict__ bzr2,
    const _Float16* __restrict__ Wg2t,  const float* __restrict__ bg2)
{
    extern __shared__ unsigned char smem[];
    _Float16* lds_x = (_Float16*)smem;                       // 32*ISZ
    _Float16* lds_h = lds_x + 32 * ISZ;                      // 32*HALF (A operand, f16)
    float*    lds_o = (float*)(lds_h + 32 * HALF);           // 32*2*HALF (GEMM out, f32)
    float*    lds_t = lds_o + 32 * 2 * HALF;                 // 32*HALF (h1n, f32)

    const int tid = threadIdx.x, nth = blockDim.x;
    const int lane = tid & 31, wave = tid >> 5, nwaves = nth >> 5;
    const int H2 = 2 * HALF;

    // Stage x_t and h2 (as f16 A operand) into LDS
    for (int i = tid; i < 32 * ISZ; i += nth) lds_x[i] = x_h[i];
    for (int i = tid; i < 32 * HALF; i += nth) {
        int r = i / HALF, c = i % HALF;
        lds_h[i] = (_Float16)h[r * H2 + HALF + c];
    }
    __syncthreads();

    // (1) zr1_pre = [x | h2] @ Wzr1 + b     -> lds_o[:, 0:H2]
    wmma_subgemm(lds_x, lds_h, ISZ, HALF, Wzr1t, bzr1, lds_o, 0, H2, lane, wave, nwaves);
    __syncthreads();

    // z1 / r1*h2
    for (int i = tid; i < 32 * H2; i += nth) {
        int r = i / H2, c = i % H2;
        float s = dev_sigmoid(lds_o[i]);
        if (c < HALF) lds_o[i] = 1.0f - MAXF + MAXF * s;                     // z1
        else lds_h[r * HALF + (c - HALF)] = (_Float16)(s * h[r * H2 + c]);   // r1*h2
    }
    __syncthreads();

    // (2) g1_pre = [x | r1*h2] @ Wg1 + b    -> lds_o[:, HALF:H2]
    wmma_subgemm(lds_x, lds_h, ISZ, HALF, Wg1t, bg1, lds_o, HALF, HALF, lane, wave, nwaves);
    __syncthreads();

    // h1n = z1*h1 + (1-z1)*tanh(g1_pre)
    for (int i = tid; i < 32 * HALF; i += nth) {
        int r = i / HALF, c = i % HALF;
        float z1  = lds_o[r * H2 + c];
        float g1  = tanhf(lds_o[r * H2 + HALF + c]);
        float h1n = z1 * h[r * H2 + c] + (1.0f - z1) * g1;
        lds_t[i] = h1n;
        lds_h[i] = (_Float16)h1n;
        y_h[r * H2 + c] = (_Float16)h1n;
    }
    __syncthreads();

    // (3) zr2_pre = [x | h1n] @ Wzr2 + b    -> lds_o[:, 0:H2]
    wmma_subgemm(lds_x, lds_h, ISZ, HALF, Wzr2t, bzr2, lds_o, 0, H2, lane, wave, nwaves);
    __syncthreads();

    // z2 / r2*h1n
    for (int i = tid; i < 32 * H2; i += nth) {
        int r = i / H2, c = i % H2;
        float s = dev_sigmoid(lds_o[i]);
        if (c < HALF) lds_o[i] = 1.0f - MAXF + MAXF * s;                        // z2
        else lds_h[r * HALF + (c - HALF)] = (_Float16)(s * lds_t[r * HALF + (c - HALF)]);
    }
    __syncthreads();

    // (4) g2_pre = [x | r2*h1n] @ Wg2 + b   -> lds_o[:, HALF:H2]
    wmma_subgemm(lds_x, lds_h, ISZ, HALF, Wg2t, bg2, lds_o, HALF, HALF, lane, wave, nwaves);
    __syncthreads();

    // h2n = z2*h2 + (1-z2)*tanh(g2_pre); commit new state
    for (int i = tid; i < 32 * HALF; i += nth) {
        int r = i / HALF, c = i % HALF;
        float z2  = lds_o[r * H2 + c];
        float g2  = tanhf(lds_o[r * H2 + HALF + c]);
        float h2n = z2 * h[r * H2 + HALF + c] + (1.0f - z2) * g2;
        h[r * H2 + HALF + c] = h2n;
        h[r * H2 + c]        = lds_t[i];          // h1n
        y_h[r * H2 + HALF + c] = (_Float16)h2n;
    }
}

// ---------------------------------------------------------------------------
// Decode: out[4096, NTOKEN] = y1_h (4096x512) @ embed_h^T (+ out_b)
// embed_h is row-major NTOKEN x 512 == exactly the N-major/K-contiguous B we need.
// WG = 256 thr (8 waves), tile 64(M) x 128(N); wave tile 32x32 = 2x2 WMMA tiles.
// B tiles (128 x 32 f16 = 8 KB) are DMA'd into LDS by the Tensor Data Mover,
// double-buffered, issued by wave 0 and pipelined with s_wait_tensorcnt.
// TDM OOB rows (beyond NTOKEN) zero-fill, handling the ragged last N block.
// A is read directly from global (L2-resident) with prefetch.
// ---------------------------------------------------------------------------
#define KTILES (IN_SIZE / 32)   // 16

__global__ __launch_bounds__(256) void decode_gemm(
    const _Float16* __restrict__ Ah,       // 4096 x 512
    const _Float16* __restrict__ Bh,       // NTOKEN x 512 (embed_h)
    const float* __restrict__ out_b,       // NTOKEN
    float* __restrict__ out)               // 4096 x NTOKEN
{
    __shared__ _Float16 Bt[2][128][32];    // double-buffered B tile (2 x 8 KB)

    const int lane  = threadIdx.x & 31;
    const int wave  = threadIdx.x >> 5;
    const int mwarp = wave >> 2, nwarp = wave & 3;
    const int Mbase = blockIdx.y * 64 + mwarp * 32;
    const int Nloc  = nwarp * 32;
    const int Nbase = blockIdx.x * 128 + Nloc;
    const int l15 = lane & 15, lhi = lane >> 4;

    const _Float16* gB = Bh + (size_t)blockIdx.x * 128 * IN_SIZE;  // tile row base
    const int rows_rem = NTOKEN - blockIdx.x * 128;

    if (wave == 0)
        tdm_load_b_tile(gB, (unsigned)(uintptr_t)&Bt[0][0][0], IN_SIZE, rows_rem);

    v8f acc[2][2] = {};

    for (int kc = 0; kc < KTILES; ++kc) {
        const int cur = kc & 1;
        const int k0  = kc * 32;
        if (wave == 0) {
            if (kc + 1 < KTILES) {
                tdm_load_b_tile(gB + (k0 + 32),
                                (unsigned)(uintptr_t)&Bt[cur ^ 1][0][0],
                                IN_SIZE - (k0 + 32), rows_rem);
                __builtin_amdgcn_s_wait_tensorcnt(1);   // oldest (cur) DMA done
            } else {
                __builtin_amdgcn_s_wait_tensorcnt(0);   // last tile done
            }
        }
        __syncthreads();                                // Bt[cur] visible to all waves

        v16h a[2], b[2];
#pragma unroll
        for (int mi = 0; mi < 2; ++mi) {
            const _Float16* ap =
                Ah + (size_t)(Mbase + mi * 16 + l15) * IN_SIZE + k0 + (lhi << 4);
            __builtin_prefetch(ap + 32, 0, 1);          // next K chunk of A
            a[mi] = ld16_pair(ap, ap + 8);
        }
#pragma unroll
        for (int nj = 0; nj < 2; ++nj) {
            const _Float16* bp = &Bt[cur][Nloc + nj * 16 + l15][lhi << 4];
            b[nj] = ld16_pair(bp, bp + 8);
        }
#pragma unroll
        for (int mi = 0; mi < 2; ++mi)
#pragma unroll
            for (int nj = 0; nj < 2; ++nj)
                acc[mi][nj] = __builtin_amdgcn_wmma_f32_16x16x32_f16(
                    false, a[mi], false, b[nj], (short)0, acc[mi][nj], false, false);

        __syncthreads();                                // all reads of Bt[cur] done
    }

#pragma unroll
    for (int mi = 0; mi < 2; ++mi) {
#pragma unroll
        for (int nj = 0; nj < 2; ++nj) {
            const int col = Nbase + nj * 16 + l15;
            if (col < NTOKEN) {
                const float bv = out_b[col];
                const int rbase = Mbase + mi * 16 + (lhi << 3);
#pragma unroll
                for (int j = 0; j < 8; ++j)
                    out[(size_t)(rbase + j) * NTOKEN + col] = acc[mi][nj][j] + bv;
            }
        }
    }
}

// ---------------------------------------------------------------------------
// Host launcher
// ---------------------------------------------------------------------------
extern "C" void kernel_launch(void* const* d_in, const int* in_sizes, int n_in,
                              void* d_out, int out_size, void* d_ws, size_t ws_size,
                              hipStream_t stream) {
    (void)in_sizes; (void)n_in; (void)out_size; (void)ws_size;

    const int*   seq   = (const int*)  d_in[0];
    const float* h0_0  = (const float*)d_in[1];
    const float* h0_1  = (const float*)d_in[2];
    const float* embed = (const float*)d_in[3];
    const float* out_b = (const float*)d_in[4];
    const float* W[8]  = { (const float*)d_in[5],  (const float*)d_in[7],
                           (const float*)d_in[9],  (const float*)d_in[11],
                           (const float*)d_in[13], (const float*)d_in[15],
                           (const float*)d_in[17], (const float*)d_in[19] };
    const float* B[8]  = { (const float*)d_in[6],  (const float*)d_in[8],
                           (const float*)d_in[10], (const float*)d_in[12],
                           (const float*)d_in[14], (const float*)d_in[16],
                           (const float*)d_in[18], (const float*)d_in[20] };
    // W dims (K, N): layer0 K=1024, layer1 K=1280
    const int WK[8] = {1024, 1024, 1024, 1024, 1280, 1280, 1280, 1280};
    const int WN[8] = {1024,  512, 1024,  512,  512,  256,  512,  256};

    // ---- workspace layout ----
    char* ws = (char*)d_ws;
    size_t off = 0;
    auto take = [&](size_t bytes) -> char* {
        char* p = ws + off;
        off = (off + bytes + 255) & ~(size_t)255;
        return p;
    };
    _Float16* embed_h = (_Float16*)take((size_t)NTOKEN * IN_SIZE * 2);
    _Float16* Wt[8];
    for (int i = 0; i < 8; ++i) Wt[i] = (_Float16*)take((size_t)WK[i] * WN[i] * 2);
    _Float16* x0h = (_Float16*)take((size_t)SEQ * BATCH * 512  * 2);
    _Float16* y0h = (_Float16*)take((size_t)SEQ * BATCH * 1024 * 2);
    _Float16* y1h = (_Float16*)take((size_t)SEQ * BATCH * 512  * 2);
    float*    hs0 = (float*)   take((size_t)BATCH * 1024 * 4);
    float*    hs1 = (float*)   take((size_t)BATCH * 512  * 4);

    // ---- prep: f16 conversion / weight transpose / gather / state init ----
    {
        int n = NTOKEN * IN_SIZE;
        cvt_f32_to_f16<<<(n + 255) / 256, 256, 0, stream>>>(embed, embed_h, n);
        for (int i = 0; i < 8; ++i) {
            int m = WK[i] * WN[i];
            transpose_f32_to_f16<<<(m + 255) / 256, 256, 0, stream>>>(W[i], Wt[i], WK[i], WN[i]);
        }
        copy_f32<<<(BATCH * 1024 + 255) / 256, 256, 0, stream>>>(h0_0, hs0, BATCH * 1024);
        copy_f32<<<(BATCH * 512  + 255) / 256, 256, 0, stream>>>(h0_1, hs1, BATCH * 512);
        gather_rows<<<SEQ * BATCH, 256, 0, stream>>>(seq, embed_h, x0h);
    }

    // ---- sequential RevGRU scan: fused one-WGP step kernels ----
    const size_t smem0 = 32 * 512 * 2 + 32 * 512 * 2 + 32 * 1024 * 4 + 32 * 512 * 4; // 256 KB
    const size_t smem1 = 32 * 1024 * 2 + 32 * 256 * 2 + 32 * 512 * 4 + 32 * 256 * 4; // 176 KB
    for (int t = 0; t < SEQ; ++t) {
        revgru_step<512, 512><<<1, 1024, smem0, stream>>>(
            x0h + (size_t)t * BATCH * 512, hs0, y0h + (size_t)t * BATCH * 1024,
            Wt[0], B[0], Wt[1], B[1], Wt[2], B[2], Wt[3], B[3]);
        revgru_step<1024, 256><<<1, 1024, smem1, stream>>>(
            y0h + (size_t)t * BATCH * 1024, hs1, y1h + (size_t)t * BATCH * 512,
            Wt[4], B[4], Wt[5], B[5], Wt[6], B[6], Wt[7], B[7]);
    }

    // ---- decode: (4096 x 512) @ embed^T + out_b ----
    dim3 grid((NTOKEN + 127) / 128, (SEQ * BATCH) / 64);
    decode_gemm<<<grid, 256, 0, stream>>>(y1h, embed_h, out_b, (float*)d_out);
}
